// MultiHeadedAttention_89721866814060
// MI455X (gfx1250) — matrix operand order repaired and checked by reading, not compiled, not run on hardware
//
#include <hip/hip_runtime.h>
#include <hip/hip_bf16.h>

// ---------------------------------------------------------------------------
// MI455X (gfx1250) GQA attention: QKV proj (+fused RoPE) -> flash attention ->
// output proj. All matmuls on v_wmma_f32_16x16x32_bf16 (wave32); all LDS tile
// staging through the Tensor Data Mover (tensor_load_to_lds + TENSORcnt) with
// double-buffered LDS so DMA overlaps WMMA issue. Workload is compute-bound
// (~100 GFLOP vs ~100 MB traffic @ 23.3 TB/s => ~5us memory floor), so the
// bf16 matrix path with f32 accumulation is the right precision/throughput
// point; fp32 VALU would be an order of magnitude slower.
// ---------------------------------------------------------------------------

typedef __attribute__((ext_vector_type(16))) __bf16 v16bf;
typedef __attribute__((ext_vector_type(8)))  float  v8f;
typedef unsigned int u32x4 __attribute__((ext_vector_type(4)));
typedef int          i32x4 __attribute__((ext_vector_type(4)));
typedef int          i32x8 __attribute__((ext_vector_type(8)));

#define S_LEN 2048
#define D_MOD 2048
#define N_HEAD 32
#define N_KV 8
#define HD 64
#define N_QKV 3072   // (32 + 2*8) * 64

// Workspace layout (bytes). Total = 48 MiB.
#define OFF_XB   ((size_t)0)
#define OFF_WQKV (OFF_XB   + (size_t)S_LEN * D_MOD * 2)
#define OFF_WO   (OFF_WQKV + (size_t)N_QKV * D_MOD * 2)
#define OFF_Q    (OFF_WO   + (size_t)D_MOD * D_MOD * 2)
#define OFF_K    (OFF_Q    + (size_t)N_HEAD * S_LEN * HD * 2)
#define OFF_V    (OFF_K    + (size_t)N_KV * S_LEN * HD * 2)
#define OFF_AO   (OFF_V    + (size_t)N_KV * S_LEN * HD * 2)

#ifndef __has_builtin
#define __has_builtin(x) 0
#endif
#if __has_builtin(__builtin_amdgcn_tensor_load_to_lds)
#define USE_TDM 1
#else
#define USE_TDM 0
#endif

// --- Tensor Data Mover: issue a 2D tile load (rows x rowElems bf16, row
// stride gStride elems) into LDS. D# per CDNA5 ISA.
#if USE_TDM
__device__ __forceinline__ void tdm_load_2d(unsigned lds_off, const void* gaddr,
                                            unsigned rowElems, unsigned rows,
                                            unsigned gStride) {
  unsigned long long ga = (unsigned long long)gaddr;
  u32x4 g0;
  g0[0] = 1u;                                   // count=1, user-mode descriptor
  g0[1] = lds_off;                              // lds_addr (bytes)
  g0[2] = (unsigned)ga;                         // global_addr[31:0]
  g0[3] = (unsigned)((ga >> 32) & 0x1FFFFFFu)   // global_addr[56:32]
        | (2u << 30);                           // type=2 ("image")
  i32x8 g1;
  g1[0] = (int)(1u << 16);                      // data_size=1 -> 2 bytes/elem
  g1[1] = (int)((rowElems & 0xFFFFu) << 16);    // tensor_dim0[15:0] @ bit48
  g1[2] = (int)((rowElems >> 16) | ((rows & 0xFFFFu) << 16)); // dim0 hi, dim1 lo
  g1[3] = (int)((rows >> 16) | ((rowElems & 0xFFFFu) << 16)); // dim1 hi, tile_dim0
  g1[4] = (int)(rows & 0xFFFFu);                // tile_dim1 (tile_dim2=0)
  g1[5] = (int)gStride;                         // tensor_dim0_stride[31:0]
  g1[6] = 0;                                    // stride hi, dim1_stride lo
  g1[7] = 0;
  i32x4 z4 = {0, 0, 0, 0};
#if defined(__clang_major__) && __clang_major__ >= 23
  i32x8 z8 = {0, 0, 0, 0, 0, 0, 0, 0};
  __builtin_amdgcn_tensor_load_to_lds(g0, g1, z4, z4, z8, 0);
#else
  __builtin_amdgcn_tensor_load_to_lds(g0, g1, z4, z4, 0);
#endif
}
#endif

// One-call tile stage: TDM (wave 0 issues; EXEC-independent, scalar operands)
// or cooperative per-lane copy fallback.
__device__ __forceinline__ void stage_tile(__bf16* lds, const __bf16* g,
                                           int rowElems, int rows, int gStride,
                                           int wave, int tid, int nthreads) {
#if USE_TDM
  (void)tid; (void)nthreads;
  if (wave == 0)
    tdm_load_2d((unsigned)(size_t)lds, g, (unsigned)rowElems, (unsigned)rows,
                (unsigned)gStride);
#else
  (void)wave;
  int rowVec = rowElems / 8;
  int nvec = rows * rowVec;
  for (int u = tid; u < nvec; u += nthreads) {
    int r = u / rowVec, c = (u % rowVec) * 8;
    *(uint4*)&lds[r * rowElems + c] = *(const uint4*)&g[(size_t)r * gStride + c];
  }
#endif
}

template <int N>
__device__ __forceinline__ void wait_tiles() {
#if USE_TDM
#if __has_builtin(__builtin_amdgcn_s_wait_tensorcnt)
  __builtin_amdgcn_s_wait_tensorcnt(N);
#else
  asm volatile("s_wait_tensorcnt %0" ::"n"(N) : "memory");
#endif
#endif
}

// --- WMMA fragment index maps (CDNA5 ISA 7.12.2, wave32) -------------------
__device__ __forceinline__ int akmap(int i, int half) {
  int v = i >> 1;
  int base = (v < 4) ? (2 * v) : (16 + 2 * (v - 4));
  return base + 8 * half + (i & 1);
}
__device__ __forceinline__ int bkmap(int i, int half) { return (i < 8) ? (i + 8 * half) : (16 + 8 * half + (i - 8)); }

__device__ __forceinline__ v16bf load_a_rowmajor(const __bf16* base, int row,
                                                 int ld, int koff, int half) {
  v16bf f;
#pragma unroll
  for (int i = 0; i < 16; ++i) f[i] = base[row * ld + koff + akmap(i, half)];
  return f;
}
__device__ __forceinline__ v16bf load_b_from_rowsT(const __bf16* base, int nrow,
                                                   int ld, int koff, int half) {
  v16bf f;
#pragma unroll
  for (int i = 0; i < 16; ++i) f[i] = base[nrow * ld + koff + bkmap(i, half)];
  return f;
}
__device__ __forceinline__ v16bf load_b_colmajor(const __bf16* base, int col,
                                                 int ld, int half) {
  v16bf f;
#pragma unroll
  for (int i = 0; i < 16; ++i) f[i] = base[bkmap(i, half) * ld + col];
  return f;
}

__device__ __forceinline__ v8f wmma_bf16(v16bf a, v16bf b, v8f c) {
  return __builtin_amdgcn_wmma_f32_16x16x32_bf16(
      false, a, false, b, (short)0, c, false, false);
}

// ---------------------------------------------------------------------------
// Kernel 1: f32 -> bf16 conversion
// ---------------------------------------------------------------------------
__global__ void cvt_bf16(const float* __restrict__ in, __bf16* __restrict__ out,
                         int n) {
  int i = blockIdx.x * 256 + threadIdx.x;
  if (i < n) out[i] = (__bf16)in[i];
}

// ---------------------------------------------------------------------------
// Kernels 2/4 GEMM core: 128 threads (4 waves), workgroup tile 128(M)x64(N),
// K-step 32, TDM double-buffered LDS. Each wave: 32x64 output = 8 WMMA/step.
// ---------------------------------------------------------------------------
#define GEMM_PROLOGUE(Aptr, Bptr)                                              \
  __shared__ __align__(16) __bf16 As[2][128][32];                              \
  __shared__ __align__(16) __bf16 Bs[2][64][32];                               \
  const int tid = threadIdx.x;                                                 \
  const int lane = tid & 31, wave = tid >> 5, half = lane >> 4, n = lane & 15; \
  const int m0 = blockIdx.x * 128, n0 = blockIdx.y * 64;                       \
  const __bf16* Ag = (Aptr) + (size_t)m0 * D_MOD;                              \
  const __bf16* Bg = (Bptr) + (size_t)n0 * D_MOD;                              \
  v8f acc[2][4] = {};                                                          \
  stage_tile(&As[0][0][0], Ag, 32, 128, D_MOD, wave, tid, 128);                \
  stage_tile(&Bs[0][0][0], Bg, 32, 64, D_MOD, wave, tid, 128);                 \
  for (int k0 = 0; k0 < D_MOD; k0 += 32) {                                     \
    const int cur = (k0 >> 5) & 1;                                             \
    const bool hasNext = (k0 + 32) < D_MOD;                                    \
    if (hasNext) {                                                             \
      stage_tile(&As[cur ^ 1][0][0], Ag + k0 + 32, 32, 128, D_MOD, wave, tid,  \
                 128);                                                         \
      stage_tile(&Bs[cur ^ 1][0][0], Bg + k0 + 32, 32, 64, D_MOD, wave, tid,   \
                 128);                                                         \
      wait_tiles<2>(); /* previous pair (cur tile) complete */                 \
    } else {                                                                   \
      wait_tiles<0>();                                                         \
    }                                                                          \
    __syncthreads();                                                           \
    v16bf a0 = load_a_rowmajor(&As[cur][wave * 32][0], n, 32, 0, half);        \
    v16bf a1 = load_a_rowmajor(&As[cur][wave * 32 + 16][0], n, 32, 0, half);   \
    _Pragma("unroll") for (int t = 0; t < 4; ++t) {                            \
      v16bf b = load_b_from_rowsT(&Bs[cur][t * 16][0], n, 32, 0, half);        \
      acc[0][t] = wmma_bf16(a0, b, acc[0][t]);                                 \
      acc[1][t] = wmma_bf16(a1, b, acc[1][t]);                                 \
    }                                                                          \
    __syncthreads(); /* all reads of cur done before it is re-filled */        \
  }

__global__ __launch_bounds__(128) void gemm_qkv_rope(
    const __bf16* __restrict__ X, const __bf16* __restrict__ W,
    const float* __restrict__ freqs, __bf16* __restrict__ Qb,
    __bf16* __restrict__ Kb, __bf16* __restrict__ Vb) {
  GEMM_PROLOGUE(X, W)

  // Epilogue: RoPE on Q/K columns (pair (c, c^1) sits in adjacent lanes of the
  // C fragment -> one shfl_xor(1)), scatter to per-head [head][seq][64] bf16.
#pragma unroll
  for (int s = 0; s < 2; ++s) {
#pragma unroll
    for (int t = 0; t < 4; ++t) {
#pragma unroll
      for (int v = 0; v < 8; ++v) {
        int row = m0 + wave * 32 + s * 16 + v + 8 * half;
        int c = n0 + t * 16 + n;
        float val = acc[s][t][v];
        float partner = __shfl_xor(val, 1);
        float o = val;
        if (c < D_MOD + N_KV * HD) {  // Q or K column -> rotate
          int dd = c & 63;
          float2 cs = ((const float2*)freqs)[row * (HD / 2) + (dd >> 1)];
          if ((c & 1) == 0) o = val * cs.x - partner * cs.y;  // x0*cos - x1*sin
          else              o = val * cs.x + partner * cs.y;  // x1*cos + x0*sin
        }
        __bf16 ob = (__bf16)o;
        if (c < D_MOD) {
          int hh = c >> 6, dd = c & 63;
          Qb[((size_t)hh * S_LEN + row) * HD + dd] = ob;
        } else if (c < D_MOD + N_KV * HD) {
          int hh = (c - D_MOD) >> 6, dd = c & 63;
          Kb[((size_t)hh * S_LEN + row) * HD + dd] = ob;
        } else {
          int hh = (c - D_MOD - N_KV * HD) >> 6, dd = c & 63;
          Vb[((size_t)hh * S_LEN + row) * HD + dd] = ob;
        }
      }
    }
  }
}

__global__ __launch_bounds__(128) void gemm_wo(const __bf16* __restrict__ A,
                                               const __bf16* __restrict__ W,
                                               float* __restrict__ out) {
  GEMM_PROLOGUE(A, W)
#pragma unroll
  for (int s = 0; s < 2; ++s) {
#pragma unroll
    for (int t = 0; t < 4; ++t) {
#pragma unroll
      for (int v = 0; v < 8; ++v) {
        int row = m0 + wave * 32 + s * 16 + v + 8 * half;
        int c = n0 + t * 16 + n;
        out[(size_t)row * D_MOD + c] = acc[s][t][v];
      }
    }
  }
}

// ---------------------------------------------------------------------------
// Kernel 3: causal flash attention per (head, 64-row q block). 4 waves x 16
// q-rows; 32-key K/V blocks TDM double-buffered in LDS; online softmax with
// 16-lane shfl row reductions (a C-tile row spans one 16-lane half exactly).
// ---------------------------------------------------------------------------
__global__ __launch_bounds__(128) void attn_kernel(
    const __bf16* __restrict__ Qb, const __bf16* __restrict__ Kb,
    const __bf16* __restrict__ Vb, __bf16* __restrict__ AO) {
  __shared__ __align__(16) __bf16 Qs[64][64];
  __shared__ __align__(16) __bf16 Ks[2][32][64];
  __shared__ __align__(16) __bf16 Vs[2][32][64];
  __shared__ __align__(16) __bf16 Ps[4][16][32];

  const int h = blockIdx.x;
  const int qb = blockIdx.y * 64;
  const int kvh = h >> 2;  // rep = N_HEAD / N_KV = 4
  const int tid = threadIdx.x;
  const int lane = tid & 31, wave = tid >> 5, half = lane >> 4, n = lane & 15;

  const __bf16* Qh = Qb + ((size_t)h * S_LEN + qb) * HD;
  const __bf16* Kh = Kb + (size_t)kvh * S_LEN * HD;
  const __bf16* Vh = Vb + (size_t)kvh * S_LEN * HD;

  stage_tile(&Qs[0][0], Qh, 64, 64, 64, wave, tid, 128);
  stage_tile(&Ks[0][0][0], Kh, 64, 32, 64, wave, tid, 128);
  stage_tile(&Vs[0][0][0], Vh, 64, 32, 64, wave, tid, 128);
  wait_tiles<0>();
  __syncthreads();
  v16bf qa0 = load_a_rowmajor(&Qs[wave * 16][0], n, 64, 0, half);
  v16bf qa1 = load_a_rowmajor(&Qs[wave * 16][0], n, 64, 32, half);

  float mi[8], li[8];
  v8f oacc[4] = {};
#pragma unroll
  for (int v = 0; v < 8; ++v) { mi[v] = -1e30f; li[v] = 0.0f; }

  const int kend = qb + 64;  // causal: uniform per workgroup
  for (int kb = 0; kb < kend; kb += 32) {
    const int cur = (kb >> 5) & 1;
    const bool hasNext = (kb + 32) < kend;
    if (hasNext) {
      stage_tile(&Ks[cur ^ 1][0][0], Kh + (size_t)(kb + 32) * HD, 64, 32, 64,
                 wave, tid, 128);
      stage_tile(&Vs[cur ^ 1][0][0], Vh + (size_t)(kb + 32) * HD, 64, 32, 64,
                 wave, tid, 128);
      wait_tiles<2>();
    } else {
      wait_tiles<0>();
    }
    __syncthreads();

    // scores: S = Q (16x64) @ K^T (64x32), two 16x16 tiles, K-dim split 32+32
    v8f s0 = {}, s1 = {};
    {
      v16bf b0 = load_b_from_rowsT(&Ks[cur][0][0], n, 64, 0, half);
      v16bf b1 = load_b_from_rowsT(&Ks[cur][0][0], n, 64, 32, half);
      s0 = wmma_bf16(qa0, b0, s0);
      s0 = wmma_bf16(qa1, b1, s0);
      v16bf b2 = load_b_from_rowsT(&Ks[cur][16][0], n, 64, 0, half);
      v16bf b3 = load_b_from_rowsT(&Ks[cur][16][0], n, 64, 32, half);
      s1 = wmma_bf16(qa0, b2, s1);
      s1 = wmma_bf16(qa1, b3, s1);
    }

    // online softmax update (row m = v + 8*half within wave's 16 rows)
#pragma unroll
    for (int v = 0; v < 8; ++v) {
      int row = qb + wave * 16 + v + 8 * half;
      float a0 = s0[v] * 0.125f;  // 1/sqrt(64)
      float a1 = s1[v] * 0.125f;
      if (kb + n > row)      a0 = -1e30f;  // causal mask
      if (kb + 16 + n > row) a1 = -1e30f;
      float mx = fmaxf(a0, a1);
#pragma unroll
      for (int off = 8; off > 0; off >>= 1) mx = fmaxf(mx, __shfl_xor(mx, off));
      float mnew = fmaxf(mi[v], mx);
      float alpha = __expf(mi[v] - mnew);
      float p0 = __expf(a0 - mnew);
      float p1 = __expf(a1 - mnew);
      float rs = p0 + p1;
#pragma unroll
      for (int off = 8; off > 0; off >>= 1) rs += __shfl_xor(rs, off);
      li[v] = li[v] * alpha + rs;
      mi[v] = mnew;
#pragma unroll
      for (int dt = 0; dt < 4; ++dt) oacc[dt][v] = oacc[dt][v] * alpha;
      // C-layout -> LDS so P can be re-read in A-fragment layout
      Ps[wave][v + 8 * half][n]      = (__bf16)p0;
      Ps[wave][v + 8 * half][16 + n] = (__bf16)p1;
    }
    // cross-lane LDS dependency within the wave: DS ops are in-order per wave;
    // one dscnt drain makes the P tile visible to all 32 lanes.
    asm volatile("s_wait_dscnt 0" ::: "memory");

    v16bf pa = load_a_rowmajor(&Ps[wave][0][0], n, 32, 0, half);
#pragma unroll
    for (int dt = 0; dt < 4; ++dt) {  // O += P (16x32) @ V (32x64)
      v16bf vb = load_b_colmajor(&Vs[cur][0][0], dt * 16 + n, 64, half);
      oacc[dt] = wmma_bf16(pa, vb, oacc[dt]);
    }
    __syncthreads();  // all reads of cur done before it is re-filled
  }

  // epilogue: normalize, write back to [seq][D] bf16 for the output proj
#pragma unroll
  for (int dt = 0; dt < 4; ++dt) {
#pragma unroll
    for (int v = 0; v < 8; ++v) {
      int row = qb + wave * 16 + v + 8 * half;
      float o = oacc[dt][v] / li[v];
      AO[(size_t)row * D_MOD + h * HD + dt * 16 + n] = (__bf16)o;
    }
  }
}

// ---------------------------------------------------------------------------
extern "C" void kernel_launch(void* const* d_in, const int* in_sizes, int n_in,
                              void* d_out, int out_size, void* d_ws,
                              size_t ws_size, hipStream_t stream) {
  const float* hs    = (const float*)d_in[0];  // [1, 2048, 2048]
  const float* freqs = (const float*)d_in[1];  // [2048, 32, 2]
  const float* wqkv  = (const float*)d_in[2];  // [3072, 2048]
  const float* wo    = (const float*)d_in[3];  // [2048, 2048]
  float* out = (float*)d_out;                  // [1, 2048, 2048]

  char* ws = (char*)d_ws;  // needs 48 MiB
  __bf16* Xb  = (__bf16*)(ws + OFF_XB);
  __bf16* Wqb = (__bf16*)(ws + OFF_WQKV);
  __bf16* Wob = (__bf16*)(ws + OFF_WO);
  __bf16* Qb  = (__bf16*)(ws + OFF_Q);
  __bf16* Kb  = (__bf16*)(ws + OFF_K);
  __bf16* Vb  = (__bf16*)(ws + OFF_V);
  __bf16* AOb = (__bf16*)(ws + OFF_AO);

  const int nX  = S_LEN * D_MOD;  // 4194304
  const int nWq = N_QKV * D_MOD;  // 6291456
  const int nWo = D_MOD * D_MOD;  // 4194304
  cvt_bf16<<<nX / 256, 256, 0, stream>>>(hs, Xb, nX);
  cvt_bf16<<<nWq / 256, 256, 0, stream>>>(wqkv, Wqb, nWq);
  cvt_bf16<<<nWo / 256, 256, 0, stream>>>(wo, Wob, nWo);

  gemm_qkv_rope<<<dim3(S_LEN / 128, N_QKV / 64), 128, 0, stream>>>(
      Xb, Wqb, freqs, Qb, Kb, Vb);

  attn_kernel<<<dim3(N_HEAD, S_LEN / 64), 128, 0, stream>>>(Qb, Kb, Vb, AOb);

  gemm_wo<<<dim3(S_LEN / 128, D_MOD / 64), 128, 0, stream>>>(AOb, Wob, out);
}
